// Attention_87428354277802
// MI455X (gfx1250) — matrix operand order repaired and checked by reading, not compiled, not run on hardware
//
#include <hip/hip_runtime.h>

// ---------------------------------------------------------------------------
// Fused QKV projection + causal flash attention for gfx1250 (MI455X).
// All matmuls run on v_wmma_f32_16x16x32_bf16 (wave32 WMMA), f32 accumulate.
// ---------------------------------------------------------------------------

#define BDIM  2
#define TSEQ  4096
#define CDIM  768
#define NHEAD 12
#define HD    64
#define N3C   (3 * CDIM)

typedef __attribute__((ext_vector_type(16))) __bf16       bf16x16;
typedef __attribute__((ext_vector_type(8)))  float        floatx8;
typedef __attribute__((ext_vector_type(4)))  float        floatx4;
typedef __attribute__((ext_vector_type(4)))  unsigned int uintx4;

// Trivial union: build/load fragments as raw bf16 bits, feed vector to WMMA.
union FragB {
    bf16x16        v;
    unsigned short u[16];
    uintx4         q[2];
};

__device__ __forceinline__ unsigned short f2bf(float x) {
    unsigned int b = __float_as_uint(x);
    b += 0x7fffu + ((b >> 16) & 1u);   // round-to-nearest-even
    return (unsigned short)(b >> 16);
}

__device__ __forceinline__ floatx8 wmma_bf16(const bf16x16& a, const bf16x16& b,
                                             const floatx8& c) {
    return __builtin_amdgcn_wmma_f32_16x16x32_bf16(false, a, false, b,
                                                   (short)0, c, false, false);
}

// ---------------------------------------------------------------------------
// Kernel 0: W [K=768][N=2304] f32  ->  Wt [N][K] bf16 (so GEMM B-fragments
// read contiguous 16B chunks).
// ---------------------------------------------------------------------------
__global__ __launch_bounds__(256) void wt_conv(const float* __restrict__ w,
                                               unsigned short* __restrict__ wT) {
    const int idx = blockIdx.x * 256 + threadIdx.x;   // 768*2304 total
    const int k   = idx / N3C;
    const int n   = idx - k * N3C;
    wT[(size_t)n * CDIM + k] = f2bf(w[idx]);
}

// ---------------------------------------------------------------------------
// Kernel 1: qkv = x @ W + bias, scattered into per-head bf16 Q/K/V buffers
// laid out [B][NH][T][64]. Block = 8 waves; wave = 16(M) x 64(N) strip.
// ---------------------------------------------------------------------------
__global__ __launch_bounds__(256) void qkv_gemm(
    const float* __restrict__ x, const unsigned short* __restrict__ wT,
    const float* __restrict__ bias,
    unsigned short* __restrict__ Qo, unsigned short* __restrict__ Ko,
    unsigned short* __restrict__ Vo) {
    const int lane = threadIdx.x & 31;
    const int wv   = threadIdx.x >> 5;
    const int col  = lane & 15;
    const int hi   = lane >> 4;
    const int n0   = blockIdx.x * 64;          // N tile (of 2304)
    const int m0   = blockIdx.y * 128 + wv * 16;  // M rows (of 8192)

    const floatx8 z8 = {0.f, 0.f, 0.f, 0.f, 0.f, 0.f, 0.f, 0.f};
    floatx8 acc[4] = {z8, z8, z8, z8};

    const float* xrow = x + (size_t)(m0 + col) * CDIM;
    const int kbA = hi * 8;   // A-frag: 8-contig k per half, halves 16 apart

    for (int k0 = 0; k0 < CDIM; k0 += 32) {
        FragB a;
        {
            const floatx4* p0 = (const floatx4*)(xrow + k0 + kbA);
            const floatx4* p1 = (const floatx4*)(xrow + k0 + kbA + 16);
            floatx4 c0 = p0[0], c1 = p0[1], c2 = p1[0], c3 = p1[1];
#pragma unroll
            for (int e = 0; e < 4; ++e) {
                a.u[e]      = f2bf(c0[e]);
                a.u[4 + e]  = f2bf(c1[e]);
                a.u[8 + e]  = f2bf(c2[e]);
                a.u[12 + e] = f2bf(c3[e]);
            }
        }
#pragma unroll
        for (int j = 0; j < 4; ++j) {
            // B-frag: lane = column n, 8-contig k per half, halves 16 apart
            const unsigned short* wrow =
                wT + (size_t)(n0 + 16 * j + col) * CDIM + k0 + hi * 8;
            FragB b;
            b.q[0] = *(const uintx4*)(wrow);
            b.q[1] = *(const uintx4*)(wrow + 16);
            acc[j] = wmma_bf16(a.v, b.v, acc[j]);
        }
    }

#pragma unroll
    for (int j = 0; j < 4; ++j) {
        const int n     = n0 + 16 * j + col;
        const float bv  = bias[n];
        const int which = n / CDIM;           // 0=Q 1=K 2=V
        const int c     = n - which * CDIM;
        const int head  = c >> 6, d = c & 63;
        unsigned short* dst = (which == 0) ? Qo : ((which == 1) ? Ko : Vo);
#pragma unroll
        for (int r = 0; r < 8; ++r) {
            const int m  = m0 + r + 8 * hi;
            const int bb = m >> 12;            // /4096
            const int t  = m & (TSEQ - 1);
            dst[(((size_t)(bb * NHEAD + head) * TSEQ + t) << 6) + d] =
                f2bf(acc[j][r] + bv);
        }
    }
}

// ---------------------------------------------------------------------------
// Kernel 2: causal flash attention. Block = 8 waves x 16 query rows = 128
// queries; streams 64-key tiles with online softmax. Scores never touch HBM.
// ---------------------------------------------------------------------------
__global__ __launch_bounds__(256) void flash_attn(
    const unsigned short* __restrict__ Qb, const unsigned short* __restrict__ Kb,
    const unsigned short* __restrict__ Vb, float* __restrict__ out) {
    __shared__ __align__(16) unsigned short sVt[64 * 72];      // V^T tile
    __shared__ __align__(16) unsigned short sP[8 * 16 * 72];   // per-wave P

    const int lane = threadIdx.x & 31;
    const int wv   = threadIdx.x >> 5;
    const int col  = lane & 15;
    const int hi   = lane >> 4;
    const int bh   = blockIdx.y;             // b*NH + h
    const int q0   = blockIdx.x * 128;
    const int qw   = q0 + wv * 16;

    const unsigned short* Qh = Qb + ((size_t)bh * TSEQ << 6);
    const unsigned short* Kh = Kb + ((size_t)bh * TSEQ << 6);
    const unsigned short* Vh = Vb + ((size_t)bh * TSEQ << 6);

    // Resident Q fragments (head dim = 64 -> two k-steps of 32)
    FragB qf[2];
    {
        const unsigned short* qrow = Qh + ((size_t)(qw + col) << 6);
#pragma unroll
        for (int ks = 0; ks < 2; ++ks) {
            const int d0 = ks * 32 + hi * 8;
            qf[ks].q[0] = *(const uintx4*)(qrow + d0);
            qf[ks].q[1] = *(const uintx4*)(qrow + d0 + 16);
        }
    }

    const floatx8 z8 = {0.f, 0.f, 0.f, 0.f, 0.f, 0.f, 0.f, 0.f};
    floatx8 oacc[4] = {z8, z8, z8, z8};
    float mrow[8], lrow[8];
#pragma unroll
    for (int r = 0; r < 8; ++r) { mrow[r] = -1e30f; lrow[r] = 0.f; }

    const int ntiles = blockIdx.x * 2 + 2;   // causal: keys <= q0+127
    unsigned short* pw = sP + wv * 16 * 72;

    for (int kt = 0; kt < ntiles; ++kt) {
        const int kv0 = kt * 64;

        // Cooperative: stage V tile transposed (Vt[d][key]) in LDS.
        {
            const int key = threadIdx.x >> 2;           // 0..63
            const int dc  = (threadIdx.x & 3) * 16;     // 0,16,32,48
            const unsigned short* vrow = Vh + ((size_t)(kv0 + key) << 6) + dc;
            uintx4 a0 = *(const uintx4*)(vrow);
            uintx4 a1 = *(const uintx4*)(vrow + 8);
            __align__(16) unsigned short tmp[16];
            *(uintx4*)(&tmp[0]) = a0;
            *(uintx4*)(&tmp[8]) = a1;
#pragma unroll
            for (int i = 0; i < 16; ++i) sVt[(dc + i) * 72 + key] = tmp[i];
        }
        __syncthreads();

        // S = Q @ K^T : K rows read directly (B-frag wants contig d per lane)
        floatx8 sacc[4] = {z8, z8, z8, z8};
#pragma unroll
        for (int ks = 0; ks < 2; ++ks) {
#pragma unroll
            for (int j = 0; j < 4; ++j) {
                const unsigned short* krow =
                    Kh + ((size_t)(kv0 + 16 * j + col) << 6) + ks * 32 + hi * 8;
                FragB b;
                b.q[0] = *(const uintx4*)(krow);
                b.q[1] = *(const uintx4*)(krow + 16);
                sacc[j] = wmma_bf16(qf[ks].v, b.v, sacc[j]);
            }
        }

        // Scale + causal mask + online softmax (16-lane butterfly per half)
#pragma unroll
        for (int r = 0; r < 8; ++r) {
            const int qi = qw + r + 8 * hi;
            float tm = -1e30f;
#pragma unroll
            for (int j = 0; j < 4; ++j) {
                const int ki = kv0 + 16 * j + col;
                float sv = sacc[j][r] * 0.125f;      // 1/sqrt(64)
                sv = (ki <= qi) ? sv : -1e30f;
                sacc[j][r] = sv;
                tm = fmaxf(tm, sv);
            }
#pragma unroll
            for (int off = 8; off >= 1; off >>= 1)
                tm = fmaxf(tm, __shfl_xor(tm, off, 32));
            const float mn = fmaxf(mrow[r], tm);
            const float al = __expf(mrow[r] - mn);
            float ts = 0.f;
#pragma unroll
            for (int j = 0; j < 4; ++j) {
                const float p = __expf(sacc[j][r] - mn);
                sacc[j][r] = p;
                ts += p;
            }
#pragma unroll
            for (int off = 8; off >= 1; off >>= 1)
                ts += __shfl_xor(ts, off, 32);
            lrow[r] = lrow[r] * al + ts;
            mrow[r] = mn;
#pragma unroll
            for (int j = 0; j < 4; ++j) oacc[j][r] *= al;
        }

        // P (C-layout) -> LDS bf16, re-read as A-fragments
#pragma unroll
        for (int j = 0; j < 4; ++j)
#pragma unroll
            for (int r = 0; r < 8; ++r)
                pw[(r + 8 * hi) * 72 + 16 * j + col] = f2bf(sacc[j][r]);

        asm volatile("s_wait_dscnt 0" ::: "memory");   // intra-wave LDS RAW

        // O += P @ V
#pragma unroll
        for (int ks = 0; ks < 2; ++ks) {
            FragB pf;
            const unsigned short* prow = pw + (size_t)col * 72 + ks * 32 + hi * 8;
            pf.q[0] = *(const uintx4*)(prow);
            pf.q[1] = *(const uintx4*)(prow + 16);
#pragma unroll
            for (int j = 0; j < 4; ++j) {
                const unsigned short* vtrow =
                    sVt + (size_t)(16 * j + col) * 72 + ks * 32 + hi * 8;
                FragB b;
                b.q[0] = *(const uintx4*)(vtrow);
                b.q[1] = *(const uintx4*)(vtrow + 16);
                oacc[j] = wmma_bf16(pf.v, b.v, oacc[j]);
            }
        }
        __syncthreads();   // protect sVt before next tile
    }

    // Epilogue: out[b][t][h*64+d] = O / l
    const int b = bh / NHEAD;
    const int h = bh - b * NHEAD;
#pragma unroll
    for (int j = 0; j < 4; ++j) {
        const int c = h * HD + 16 * j + col;
#pragma unroll
        for (int r = 0; r < 8; ++r) {
            const int t = qw + r + 8 * hi;
            out[(size_t)(b * TSEQ + t) * CDIM + c] = oacc[j][r] / lrow[r];
        }
    }
}

// ---------------------------------------------------------------------------
extern "C" void kernel_launch(void* const* d_in, const int* in_sizes, int n_in,
                              void* d_out, int out_size, void* d_ws, size_t ws_size,
                              hipStream_t stream) {
    (void)in_sizes; (void)n_in; (void)out_size; (void)ws_size;
    const float* x    = (const float*)d_in[0];
    const float* w    = (const float*)d_in[1];
    const float* bias = (const float*)d_in[2];
    // d_in[3] = n_head (device scalar) — fixed 12 at compile time.

    char* ws = (char*)d_ws;
    const size_t wt_bytes  = (size_t)N3C * CDIM * 2;             // 3.54 MB
    const size_t qkv_bytes = (size_t)BDIM * NHEAD * TSEQ * HD * 2; // 12.6 MB
    unsigned short* wT = (unsigned short*)ws;
    unsigned short* Qb = (unsigned short*)(ws + wt_bytes);
    unsigned short* Kb = (unsigned short*)(ws + wt_bytes + qkv_bytes);
    unsigned short* Vb = (unsigned short*)(ws + wt_bytes + 2 * qkv_bytes);
    float* out = (float*)d_out;

    wt_conv<<<(768 * N3C) / 256, 256, 0, stream>>>(w, wT);
    qkv_gemm<<<dim3(N3C / 64, (BDIM * TSEQ) / 128), 256, 0, stream>>>(
        x, wT, bias, Qb, Kb, Vb);
    flash_attn<<<dim3(TSEQ / 128, BDIM * NHEAD), 256, 0, stream>>>(
        Qb, Kb, Vb, out);
}
